// Encoder_13271448945119
// MI455X (gfx1250) — matrix operand order repaired and checked by reading, not compiled, not run on hardware
//
#include <hip/hip_runtime.h>
#include <hip/hip_bf16.h>

// ---------------- problem dims ----------------
#define BDIM 8
#define SDIM 1024
#define DDIM 1024
#define HDIM 16
#define HSZ  64
#define MTOK (BDIM * SDIM)          // 8192 tokens
#define QKN  (HDIM * 2 * HSZ)       // 2048
#define NEG_INF_F (-1e9f)
#define EPS_F 1e-6f
#define LAMBDA_INIT_F 0.8f

// ---------------- types ----------------
typedef unsigned short bf16s;  // raw bf16 storage
typedef __bf16 v16bf __attribute__((ext_vector_type(16)));
typedef float  v8f   __attribute__((ext_vector_type(8)));
typedef unsigned int u32x4 __attribute__((ext_vector_type(4)));

__device__ __forceinline__ bf16s f2bf(float f) {
    unsigned int u = __float_as_uint(f);
    unsigned int r = (u + 0x7fffu + ((u >> 16) & 1u)) >> 16;   // RNE
    return (bf16s)r;
}

// Load 16 bf16 elements from two 16B-aligned chunks into a WMMA operand.
__device__ __forceinline__ v16bf ld2(const bf16s* p0, const bf16s* p1) {
    union { v16bf v; u32x4 u[2]; } r;
    r.u[0] = *reinterpret_cast<const u32x4*>(p0);
    r.u[1] = *reinterpret_cast<const u32x4*>(p1);
    return r.v;
}

__device__ __forceinline__ v8f wmma_bf16(v16bf a, v16bf b, v8f c) {
    return __builtin_amdgcn_wmma_f32_16x16x32_bf16(false, a, false, b,
                                                   (short)0, c, false, false);
}

// CDNA5 async copy: 16B/lane global -> LDS, tracked by ASYNCcnt.
__device__ __forceinline__ void async_ld16(unsigned lds_off, const bf16s* gptr) {
    asm volatile("global_load_async_to_lds_b128 %0, %1, off"
                 :: "v"(lds_off), "v"((unsigned long long)(uintptr_t)gptr)
                 : "memory");
}
__device__ __forceinline__ void wait_async0() {
    asm volatile("s_wait_asynccnt 0x0" ::: "memory");
}

#define V8F_ZERO {0.f,0.f,0.f,0.f,0.f,0.f,0.f,0.f}

// ---------------- weight transpose-cast: Wt[n*K+k] = bf16(W[k*N+n]) ----------------
__global__ void wcast_kernel(const float* __restrict__ W, bf16s* __restrict__ Wt,
                             int K, int N) {
    long idx = (long)blockIdx.x * blockDim.x + threadIdx.x;
    long tot = (long)K * N;
    if (idx >= tot) return;
    int k = (int)(idx / N), n = (int)(idx % N);
    Wt[(long)n * K + k] = f2bf(W[idx]);
}

// ---------------- per-head lambda ----------------
__global__ void lambda_kernel(const float* __restrict__ lq1, const float* __restrict__ lk1,
                              const float* __restrict__ lq2, const float* __restrict__ lk2,
                              float* __restrict__ lam) {
    int h = threadIdx.x;
    if (h < HDIM) {
        float s1 = 0.f, s2 = 0.f;
        for (int i = 0; i < HSZ; ++i) {
            s1 += lq1[h * HSZ + i] * lk1[h * HSZ + i];
            s2 += lq2[h * HSZ + i] * lk2[h * HSZ + i];
        }
        lam[h] = __expf(s1) - __expf(s2) + LAMBDA_INIT_F;
    }
}

// ---------------- RMSNorm (one block per row of D=1024) ----------------
__global__ __launch_bounds__(256)
void rmsnorm_kernel(const float* __restrict__ x, const float* __restrict__ g,
                    float* __restrict__ of, bf16s* __restrict__ ob) {
    __shared__ float red[256];
    const int row = blockIdx.x;
    const int tid = threadIdx.x;
    const float* xr = x + (long)row * DDIM;
    float vals[4];
    float s = 0.f;
    for (int i = 0; i < 4; ++i) { float v = xr[tid + i * 256]; vals[i] = v; s += v * v; }
    red[tid] = s; __syncthreads();
    for (int st = 128; st > 0; st >>= 1) { if (tid < st) red[tid] += red[tid + st]; __syncthreads(); }
    const float scale = rsqrtf(red[0] / (float)DDIM + EPS_F);
    for (int i = 0; i < 4; ++i) {
        int c = tid + i * 256;
        float hv = vals[i] * scale * g[c];
        if (of) of[(long)row * DDIM + c] = hv;
        ob[(long)row * DDIM + c] = f2bf(hv);
    }
}

// ---------------- bf16 WMMA GEMM, async-LDS double-buffered ----------------
// C[M,N] = A[M,K] * Bt[N,K]^T.  Block tile 64x128 (8 waves: 4M x 2N),
// wave tile 16x64 (4 accumulators sharing one A operand).
// Per k-step (32): 12KB staged into LDS via global_load_async_to_lds_b128.
// MODE 0: Cf = acc   MODE 1: Cf = acc + R   MODE 2: Cb = bf16(acc)
#define GEMM_BUF_BYTES 12288    // A: 64*64B = 4KB, B: 128*64B = 8KB
template <int MODE>
__global__ __launch_bounds__(256)
void gemm_bf16_kernel(const bf16s* __restrict__ A, const bf16s* __restrict__ Bt,
                      float* __restrict__ Cf, bf16s* __restrict__ Cb,
                      const float* __restrict__ R, int N, int K) {
    __shared__ char lds[2 * GEMM_BUF_BYTES];
    const int tid  = threadIdx.x;
    const int lane = tid & 31;
    const int wave = tid >> 5;
    const int wm   = wave & 3;          // wave M index (0..3)
    const int wn   = wave >> 2;         // wave N index (0..1)
    const int m0   = blockIdx.y * 64;
    const int n0   = blockIdx.x * 128;

    // staging: chunk = 16B. A: 256 chunks (row=c>>2, part=c&3). B: 512 chunks.
    const long  arowG = (long)(m0 + (tid >> 2)) * K + (tid & 3) * 8;
    const long  browG0 = (long)(n0 + (tid >> 2)) * K + (tid & 3) * 8;
    const long  browG1 = (long)(n0 + ((tid + 256) >> 2)) * K + (tid & 3) * 8;
    const unsigned aoffL  = (unsigned)(tid * 16);
    const unsigned boffL0 = (unsigned)(4096 + tid * 16);
    const unsigned boffL1 = (unsigned)(4096 + (tid + 256) * 16);

    // compute-phase LDS addressing
    const int mloc  = wm * 16 + (lane & 15);        // row in 64-row A tile
    const int half  = lane >> 4;
    const int aByte = mloc * 64 + half * 16;        // A elems 0..7 ; +32 -> elems 8..15
    const int nbase = wn * 64;

    v8f acc[4] = {V8F_ZERO, V8F_ZERO, V8F_ZERO, V8F_ZERO};

    // prologue: stage buffer 0
    {
        unsigned base = 0;
        async_ld16(base + aoffL,  A  + arowG);
        async_ld16(base + boffL0, Bt + browG0);
        async_ld16(base + boffL1, Bt + browG1);
    }
    wait_async0();
    __syncthreads();

    const int nsteps = K / 32;
    for (int s = 0; s < nsteps; ++s) {
        const int cur = s & 1;
        if (s + 1 < nsteps) {                         // stage next buffer
            unsigned base = (unsigned)((cur ^ 1) * GEMM_BUF_BYTES);
            int k0n = (s + 1) * 32;
            async_ld16(base + aoffL,  A  + arowG  + k0n);
            async_ld16(base + boffL0, Bt + browG0 + k0n);
            async_ld16(base + boffL1, Bt + browG1 + k0n);
        }
        const char* ab = lds + cur * GEMM_BUF_BYTES;
        const char* bb = ab + 4096;
        union { v16bf v; u32x4 u[2]; } av;
        av.u[0] = *reinterpret_cast<const u32x4*>(ab + aByte);
        av.u[1] = *reinterpret_cast<const u32x4*>(ab + aByte + 32);
        for (int j = 0; j < 4; ++j) {
            int n = nbase + j * 16 + (lane & 15);
            union { v16bf v; u32x4 u[2]; } bv;
            bv.u[0] = *reinterpret_cast<const u32x4*>(bb + n * 64 + half * 16);
            bv.u[1] = *reinterpret_cast<const u32x4*>(bb + n * 64 + half * 16 + 32);
            acc[j] = wmma_bf16(av.v, bv.v, acc[j]);
        }
        wait_async0();          // own 3 async ops for next buffer complete
        __syncthreads();        // all waves done reading cur / staging done
    }

    // epilogue: C/D layout: lanes 0-15 (M=r,N=lane), lanes 16-31 (M=8+r,N=lane-16)
    for (int j = 0; j < 4; ++j) {
        for (int r = 0; r < 8; ++r) {
            int mm = m0 + wm * 16 + r + half * 8;
            int nn = n0 + nbase + j * 16 + (lane & 15);
            long idx = (long)mm * N + nn;
            float v = acc[j][r];
            if (MODE == 2)      Cb[idx] = f2bf(v);
            else if (MODE == 1) Cf[idx] = v + R[idx];
            else                Cf[idx] = v;
        }
    }
}

// ---------------- V transpose: [B,S,H,HS] -> [B,H,HS,S] ----------------
__global__ void vtrans_kernel(const bf16s* __restrict__ v, bf16s* __restrict__ vt) {
    long idx = (long)blockIdx.x * blockDim.x + threadIdx.x;
    const long tot = (long)BDIM * SDIM * HDIM * HSZ;
    if (idx >= tot) return;
    int e = (int)(idx % HSZ); long t = idx / HSZ;
    int h = (int)(t % HDIM);  t /= HDIM;
    int s = (int)(t % SDIM);  int b = (int)(t / SDIM);
    vt[(((long)b * HDIM + h) * HSZ + e) * SDIM + s] = v[idx];
}

// ---------------- fused differential attention ----------------
// 1 wave per block; block = (b, h, 16-query tile). Scores strip lives in LDS.
__global__ __launch_bounds__(32)
void diff_attn_kernel(const bf16s* __restrict__ q, const bf16s* __restrict__ k,
                      const bf16s* __restrict__ vt, const unsigned char* __restrict__ mask,
                      const float* __restrict__ lam, bf16s* __restrict__ o) {
    extern __shared__ char smem[];
    float* sc = (float*)smem;                                      // [2][16][SDIM] fp32
    bf16s* at = (bf16s*)(smem + 2 * 16 * SDIM * sizeof(float));    // [16][SDIM] bf16

    const int NT = SDIM / 16;
    const int wg = blockIdx.x;
    const int qt = wg % NT;
    const int h  = (wg / NT) % HDIM;
    const int b  = wg / (NT * HDIM);
    const int lane = threadIdx.x;
    const int m    = lane & 15;
    const int half = lane >> 4;
    const int kbA  = half * 8;
    const int kbB  = half * 16;

    // ---- load Q A-operands once: [2 comps][2 K-chunks of 32] ----
    const long qbase = (((long)b * SDIM + qt * 16 + m) * HDIM + h) * 2 * HSZ;
    v16bf qa[2][2];
    for (int c = 0; c < 2; ++c)
        for (int kc = 0; kc < 2; ++kc)
            qa[c][kc] = ld2(q + qbase + c * HSZ + kc * 32 + kbA,
                            q + qbase + c * HSZ + kc * 32 + kbA + 16);

    // ---- score strip: 64 key tiles x 2 components, 2 chained WMMAs each ----
    for (int kt = 0; kt < NT; ++kt) {
        const long kbase = (((long)b * SDIM + kt * 16 + m) * HDIM + h) * 2 * HSZ;
        for (int c = 0; c < 2; ++c) {
            v8f acc = V8F_ZERO;
            v16bf b0 = ld2(k + kbase + c * HSZ + kbA,      k + kbase + c * HSZ + kbA + 16);
            acc = wmma_bf16(qa[c][0], b0, acc);
            v16bf b1 = ld2(k + kbase + c * HSZ + 32 + kbA, k + kbase + c * HSZ + 32 + kbA + 16);
            acc = wmma_bf16(qa[c][1], b1, acc);
            for (int r = 0; r < 8; ++r) {
                int row = r + half * 8;
                int col = kt * 16 + (lane & 15);
                sc[(c * 16 + row) * SDIM + col] = acc[r];
            }
        }
    }
    __syncthreads();

    // ---- masked softmax: 32 lanes <-> 32 rows (2 comps x 16 rows) ----
    {
        float* row = sc + (half * 16 + m) * SDIM;
        const unsigned char* mrow = mask + (long)b * SDIM;
        float mx = -3.4e38f;
        for (int t = 0; t < SDIM; ++t) {
            float v = row[t] * 0.125f;                 // 1/sqrt(HS)
            v = mrow[t] ? v : NEG_INF_F;
            row[t] = v;
            mx = fmaxf(mx, v);
        }
        float sum = 0.f;
        for (int t = 0; t < SDIM; ++t) { float e = __expf(row[t] - mx); row[t] = e; sum += e; }
        float inv = 1.f / sum;
        for (int t = 0; t < SDIM; ++t) row[t] *= inv;
    }
    __syncthreads();

    // ---- combine: attn = a0 - lam[h]*a1 -> bf16 strip ----
    if (half == 0) {
        const float lm = lam[h];
        const float* p1 = sc + m * SDIM;
        const float* p2 = sc + (16 + m) * SDIM;
        bf16s* ar = at + m * SDIM;
        for (int t = 0; t < SDIM; ++t) ar[t] = f2bf(p1[t] - lm * p2[t]);
    }
    __syncthreads();

    // ---- PV: O[16, HS] = attn[16, S] * Vt[HS, S]^T ----
    const long vtb = ((long)b * HDIM + h) * HSZ * SDIM;
    for (int nt = 0; nt < HSZ / 16; ++nt) {
        v8f acc = V8F_ZERO;
        const long vrow = vtb + (long)(nt * 16 + m) * SDIM;
        const bf16s* arow = at + m * SDIM;
        for (int kk = 0; kk < SDIM / 32; ++kk) {
            v16bf a = ld2(arow + kk * 32 + kbA, arow + kk * 32 + kbA + 16);
            v16bf bv = ld2(vt + vrow + kk * 32 + kbA, vt + vrow + kk * 32 + kbA + 16);
            acc = wmma_bf16(a, bv, acc);
        }
        for (int r = 0; r < 8; ++r) {
            int row = qt * 16 + r + half * 8;
            int col = h * HSZ + nt * 16 + (lane & 15);
            o[((long)b * SDIM + row) * (HDIM * HSZ) + col] = f2bf(acc[r]);
        }
    }
}

// ---------------- SwiGLU elementwise: act = bf16(silu(G) * U) ----------------
__global__ void silu_mul_kernel(const float* __restrict__ G, const float* __restrict__ U,
                                bf16s* __restrict__ act, long n) {
    long i = (long)blockIdx.x * blockDim.x + threadIdx.x;
    if (i < n) {
        float g = G[i];
        float s = g / (1.f + __expf(-g));
        act[i] = f2bf(s * U[i]);
    }
}

// ---------------- host launch ----------------
extern "C" void kernel_launch(void* const* d_in, const int* in_sizes, int n_in,
                              void* d_out, int out_size, void* d_ws, size_t ws_size,
                              hipStream_t stream) {
    const float* x   = (const float*)d_in[0];
    const float* g1  = (const float*)d_in[1];
    const float* g2  = (const float*)d_in[2];
    const float* Wq  = (const float*)d_in[3];
    const float* Wk  = (const float*)d_in[4];
    const float* Wv  = (const float*)d_in[5];
    const float* Wo  = (const float*)d_in[6];
    const float* lq1 = (const float*)d_in[7];
    const float* lk1 = (const float*)d_in[8];
    const float* lq2 = (const float*)d_in[9];
    const float* lk2 = (const float*)d_in[10];
    const float* Wg  = (const float*)d_in[11];
    const float* Wu  = (const float*)d_in[12];
    const float* Wd  = (const float*)d_in[13];
    const unsigned char* pmask = (const unsigned char*)d_in[14];
    float* out = (float*)d_out;

    // ---- workspace carve-out (256B aligned) ----
    char* base = (char*)d_ws;
    size_t off = 0;
    auto alloc = [&](size_t bytes) -> char* {
        char* p = base + off;
        off += (bytes + 255) & ~(size_t)255;
        return p;
    };
    bf16s* wq_t = (bf16s*)alloc((size_t)DDIM * QKN * 2);
    bf16s* wk_t = (bf16s*)alloc((size_t)DDIM * QKN * 2);
    bf16s* wv_t = (bf16s*)alloc((size_t)DDIM * DDIM * 2);
    bf16s* wo_t = (bf16s*)alloc((size_t)DDIM * DDIM * 2);
    bf16s* wg_t = (bf16s*)alloc((size_t)DDIM * DDIM * 2);
    bf16s* wu_t = (bf16s*)alloc((size_t)DDIM * DDIM * 2);
    bf16s* wd_t = (bf16s*)alloc((size_t)DDIM * DDIM * 2);
    float* lam  = (float*)alloc(64 * sizeof(float));
    float* h_f  = (float*)alloc((size_t)MTOK * DDIM * 4);
    bf16s* h_b  = (bf16s*)alloc((size_t)MTOK * DDIM * 2);
    bf16s* q_b  = (bf16s*)alloc((size_t)MTOK * QKN * 2);
    bf16s* k_b  = (bf16s*)alloc((size_t)MTOK * QKN * 2);
    bf16s* v_b  = (bf16s*)alloc((size_t)MTOK * DDIM * 2);
    bf16s* vt_b = (bf16s*)alloc((size_t)MTOK * DDIM * 2);
    bf16s* o_b  = (bf16s*)alloc((size_t)MTOK * DDIM * 2);
    float* out1 = (float*)alloc((size_t)MTOK * DDIM * 4);
    bf16s* h2_b = (bf16s*)alloc((size_t)MTOK * DDIM * 2);
    // aliases onto buffers dead after attention:
    float* Gbuf = (float*)q_b;    // 32MB region, need 32MB
    float* Ubuf = (float*)k_b;
    bf16s* actb = v_b;

    const long nBig  = (long)DDIM * QKN;   // 2M
    const long nSq   = (long)DDIM * DDIM;  // 1M
    const long nTok  = (long)MTOK * DDIM;  // 8M

    // 1. weight casts (transposed) + lambda
    wcast_kernel<<<dim3((nBig + 255) / 256), 256, 0, stream>>>(Wq, wq_t, DDIM, QKN);
    wcast_kernel<<<dim3((nBig + 255) / 256), 256, 0, stream>>>(Wk, wk_t, DDIM, QKN);
    wcast_kernel<<<dim3((nSq + 255) / 256), 256, 0, stream>>>(Wv, wv_t, DDIM, DDIM);
    wcast_kernel<<<dim3((nSq + 255) / 256), 256, 0, stream>>>(Wo, wo_t, DDIM, DDIM);
    wcast_kernel<<<dim3((nSq + 255) / 256), 256, 0, stream>>>(Wg, wg_t, DDIM, DDIM);
    wcast_kernel<<<dim3((nSq + 255) / 256), 256, 0, stream>>>(Wu, wu_t, DDIM, DDIM);
    wcast_kernel<<<dim3((nSq + 255) / 256), 256, 0, stream>>>(Wd, wd_t, DDIM, DDIM);
    lambda_kernel<<<1, 32, 0, stream>>>(lq1, lk1, lq2, lk2, lam);

    // 2. rmsnorm1 -> fp32 residual + bf16 activation
    rmsnorm_kernel<<<MTOK, 256, 0, stream>>>(x, g1, h_f, h_b);

    // 3. QKV projections (bf16 out); grid = (N/128, M/64)
    gemm_bf16_kernel<2><<<dim3(QKN / 128, MTOK / 64), 256, 0, stream>>>(h_b, wq_t, nullptr, q_b, nullptr, QKN, DDIM);
    gemm_bf16_kernel<2><<<dim3(QKN / 128, MTOK / 64), 256, 0, stream>>>(h_b, wk_t, nullptr, k_b, nullptr, QKN, DDIM);
    gemm_bf16_kernel<2><<<dim3(DDIM / 128, MTOK / 64), 256, 0, stream>>>(h_b, wv_t, nullptr, v_b, nullptr, DDIM, DDIM);
    vtrans_kernel<<<dim3((nTok + 255) / 256), 256, 0, stream>>>(v_b, vt_b);

    // 4. fused differential attention
    const int attn_blocks = BDIM * HDIM * (SDIM / 16);         // 8192
    const size_t attn_smem = 2 * 16 * SDIM * sizeof(float) + 16 * SDIM * sizeof(bf16s); // 160KB
    diff_attn_kernel<<<attn_blocks, 32, attn_smem, stream>>>(q_b, k_b, vt_b, pmask, lam, o_b);

    // 5. output projection + residual (out1 = o @ Wo + h)
    gemm_bf16_kernel<1><<<dim3(DDIM / 128, MTOK / 64), 256, 0, stream>>>(o_b, wo_t, out1, nullptr, h_f, DDIM, DDIM);

    // 6. rmsnorm2 -> bf16
    rmsnorm_kernel<<<MTOK, 256, 0, stream>>>(out1, g2, nullptr, h2_b);

    // 7. SwiGLU: G, U, silu-mul, down-proj + residual into d_out
    gemm_bf16_kernel<0><<<dim3(DDIM / 128, MTOK / 64), 256, 0, stream>>>(h2_b, wg_t, Gbuf, nullptr, nullptr, DDIM, DDIM);
    gemm_bf16_kernel<0><<<dim3(DDIM / 128, MTOK / 64), 256, 0, stream>>>(h2_b, wu_t, Ubuf, nullptr, nullptr, DDIM, DDIM);
    silu_mul_kernel<<<dim3((nTok + 255) / 256), 256, 0, stream>>>(Gbuf, Ubuf, actb, nTok);
    gemm_bf16_kernel<1><<<dim3(DDIM / 128, MTOK / 64), 256, 0, stream>>>(actb, wd_t, out, nullptr, out1, DDIM, DDIM);
}
